// LinearTimeSelfAttention_7559142441247
// MI455X (gfx1250) — matrix operand order repaired and checked by reading, not compiled, not run on hardware
//
#include <hip/hip_runtime.h>
#include <hip/hip_bf16.h>

typedef __attribute__((ext_vector_type(16))) __bf16 v16bf;
typedef __attribute__((ext_vector_type(8)))  __bf16 v8bf;
typedef __attribute__((ext_vector_type(8)))  float  v8f;

#define NPIX 147456              // 384*384
#define LDS_PITCH 136            // 128 + 8 bf16 pad -> 272B rows

union FragB { uint4 u[2]; v16bf v; };
union Half8 { uint4 u; v8bf v; };

__device__ __forceinline__ unsigned f2ord(float f) {
  unsigned u = __float_as_uint(f);
  return (u & 0x80000000u) ? ~u : (u | 0x80000000u);
}
__device__ __forceinline__ float ord2f(unsigned u) {
  unsigned b = (u & 0x80000000u) ? (u & 0x7fffffffu) : ~u;
  return __uint_as_float(b);
}

// wave32 butterfly ops via ds_swizzle (group-of-32: and=0x1f, or=0, xor=MASK)
template <int MASK>
__device__ __forceinline__ float swz_max(float v) {
  int s = __builtin_amdgcn_ds_swizzle(__float_as_int(v), (MASK << 10) | 0x1f);
  return fmaxf(v, __int_as_float(s));
}
template <int MASK>
__device__ __forceinline__ float swz_add(float v) {
  int s = __builtin_amdgcn_ds_swizzle(__float_as_int(v), (MASK << 10) | 0x1f);
  return v + __int_as_float(s);
}

// ---------------- Kernel 0: init accumulators + pack W_qkv into bf16 A-fragment layout
// A-frag layout (16x32 bf16): lane<16 holds K {0..15}, lane>=16 holds K {16..31}, M = lane%16
__global__ void k_init_pack(const float* __restrict__ w_qkv,
                            __bf16* __restrict__ apack,
                            unsigned* __restrict__ kmax,
                            float* __restrict__ ksum,
                            float* __restrict__ ctx) {
  int t = blockIdx.x * blockDim.x + threadIdx.x;   // 49152 threads
  if (t < 128) { kmax[t] = 0u; ksum[t] = 0.0f; }
  if (t < 4096) ctx[t] = 0.0f;
  int i = t & 15, lane = (t >> 4) & 31, frag = t >> 9;
  int kc = frag & 3, mt = frag >> 2;               // 24 mt tiles * 4 k-chunks
  int o = mt * 16 + (lane & 15);
  int c = kc * 32 + i + ((lane < 16) ? 0 : 16);
  apack[t] = (__bf16)w_qkv[o * 128 + c];
}

// ---------------- Kernel 1: qkv = W_qkv @ X + b  (bf16 WMMA, fused k-row max)
__global__ __launch_bounds__(256) void k_qkv(const float* __restrict__ x,
                                             const __bf16* __restrict__ apack,
                                             const float* __restrict__ b_qkv,
                                             __bf16* __restrict__ qkv,
                                             unsigned* __restrict__ kmax) {
  __shared__ __bf16 sB[64 * LDS_PITCH];
  const int nbase = blockIdx.x * 64;
  const int t = threadIdx.x;
  // Stage X tile 128(k) x 64(n) as LDS[n][k] bf16 (transpose on store)
  #pragma unroll
  for (int it = 0; it < 32; ++it) {
    int idx = t + it * 256;
    int k = idx >> 6, j = idx & 63;
    sB[j * LDS_PITCH + k] = (__bf16)x[k * NPIX + nbase + j];
  }
  __syncthreads();

  const int w = t >> 5, l = t & 31, l16 = l & 15;
  const int nsub = w & 3, grp = w >> 2;
  const int half = (l < 16) ? 0 : 1;

  FragB bfrag[4];
  const __bf16* bp = &sB[(nsub * 16 + l16) * LDS_PITCH + half * 16];
  #pragma unroll
  for (int kc = 0; kc < 4; ++kc) {
    const uint4* p = (const uint4*)(bp + kc * 32);
    bfrag[kc].u[0] = p[0];
    bfrag[kc].u[1] = p[1];
  }
  const int col = nbase + nsub * 16 + l16;

  for (int i = 0; i < 12; ++i) {                // 24 M-tiles split over 2 wave groups
    int mt = i * 2 + grp;
    v8f acc = {};
    #pragma unroll
    for (int kc = 0; kc < 4; ++kc) {
      FragB af;
      const uint4* p = (const uint4*)(apack + (mt * 4 + kc) * 512 + l * 16);
      af.u[0] = p[0]; af.u[1] = p[1];
      acc = __builtin_amdgcn_wmma_f32_16x16x32_bf16(false, af.v, false, bfrag[kc].v,
                                                    (short)0, acc, false, false);
    }
    int m0 = mt * 16;
    bool isK = (m0 >= 128) && (m0 < 256);
    #pragma unroll
    for (int r = 0; r < 8; ++r) {
      int row = m0 + r + half * 8;
      float vv = acc[r] + b_qkv[row];
      qkv[row * NPIX + col] = (__bf16)vv;
      if (isK) {
        float m = vv;
        m = swz_max<1>(m);
        m = swz_max<2>(m);
        m = swz_max<4>(m);
        m = swz_max<8>(m);
        if (l16 == 0) atomicMax(&kmax[row - 128], f2ord(m));
      }
    }
  }
}

// ---------------- Kernel 2: unnormalized context_h += E_h @ V_h^T ; ksum += row sums of E
__global__ __launch_bounds__(128) void k_ctx(const __bf16* __restrict__ qkv,
                                             const unsigned* __restrict__ kmax,
                                             float* __restrict__ ksum,
                                             float* __restrict__ ctx) {
  const int t = threadIdx.x;
  const int l = t & 31, l16 = l & 15;
  const int half = (l < 16) ? 0 : 1;
  const int wave = blockIdx.x * 4 + (t >> 5);
  const int NW = 288 * 4;                        // 1152 waves; 4608 chunks / 1152 = 4 each

  float mx[8];
  #pragma unroll
  for (int h = 0; h < 4; ++h)
    #pragma unroll
    for (int dt = 0; dt < 2; ++dt)
      mx[h * 2 + dt] = ord2f(kmax[h * 32 + dt * 16 + l16]);

  v8f acc[16];
  #pragma unroll
  for (int i = 0; i < 16; ++i) { v8f z = {}; acc[i] = z; }
  float sacc[8] = {0, 0, 0, 0, 0, 0, 0, 0};

  for (int c = wave; c < 4608; c += NW) {
    int base = c * 32;
    #pragma unroll
    for (int h = 0; h < 4; ++h) {
      FragB af[2];
      #pragma unroll
      for (int dt = 0; dt < 2; ++dt) {
        int row = 128 + h * 32 + dt * 16 + l16;  // k-row
        const __bf16* kp = qkv + row * NPIX + base + half * 8;
        Half8 r0, r1;
        r0.u = *(const uint4*)kp;
        r1.u = *(const uint4*)(kp + 16);
        float m = mx[h * 2 + dt];
        float s = 0.f;
        #pragma unroll
        for (int i = 0; i < 8; ++i) {
          float e0 = __expf((float)r0.v[i] - m);
          float e1 = __expf((float)r1.v[i] - m);
          s += e0 + e1;
          af[dt].v[i]     = (__bf16)e0;
          af[dt].v[i + 8] = (__bf16)e1;
        }
        sacc[h * 2 + dt] += s;
      }
      #pragma unroll
      for (int et = 0; et < 2; ++et) {
        int row = 256 + h * 32 + et * 16 + l16;  // v-row
        const __bf16* vp = qkv + row * NPIX + base + half * 8;
        FragB bf_;
        bf_.u[0] = *(const uint4*)vp;
        bf_.u[1] = *(const uint4*)(vp + 16);
        #pragma unroll
        for (int dt = 0; dt < 2; ++dt) {
          int ai = h * 4 + dt * 2 + et;
          acc[ai] = __builtin_amdgcn_wmma_f32_16x16x32_bf16(false, af[dt].v, false, bf_.v,
                                                            (short)0, acc[ai], false, false);
        }
      }
    }
  }
  // row sum-of-exp: combine lane halves (both halves hold the same rows for A frags)
  #pragma unroll
  for (int i = 0; i < 8; ++i) {
    float s = swz_add<16>(sacc[i]);
    if (half == 0) {
      int h = i >> 1, dt = i & 1;
      atomicAdd(&ksum[h * 32 + dt * 16 + l16], s);
    }
  }
  #pragma unroll
  for (int h = 0; h < 4; ++h)
    #pragma unroll
    for (int dt = 0; dt < 2; ++dt)
      #pragma unroll
      for (int et = 0; et < 2; ++et) {
        int ai = h * 4 + dt * 2 + et;
        #pragma unroll
        for (int r = 0; r < 8; ++r) {
          int d = dt * 16 + r + half * 8;
          int e = et * 16 + l16;
          atomicAdd(&ctx[h * 1024 + d * 32 + e], acc[ai][r]);
        }
      }
}

// ---------------- Kernel 3: M = W_out * blockdiag(ctx^T / rowsum), packed bf16 A-frag layout
__global__ void k_m(const float* __restrict__ w_out,
                    const float* __restrict__ ctx,
                    const float* __restrict__ ksum,
                    __bf16* __restrict__ mpack) {
  int t = blockIdx.x * blockDim.x + threadIdx.x;  // 16384 threads
  int o = t >> 7, c = t & 127;
  int h = c >> 5, d = c & 31;
  float s = 0.f;
  #pragma unroll
  for (int e = 0; e < 32; ++e)
    s += w_out[o * 128 + h * 32 + e] * ctx[h * 1024 + d * 32 + e];
  s /= ksum[c];
  int mt = o >> 4, kc = c >> 5, cc = c & 31;
  int hb = cc >> 4;
  int i = cc & 15;
  int lane = hb * 16 + (o & 15);
  mpack[((mt * 4 + kc) * 32 + lane) * 16 + i] = (__bf16)s;
}

// ---------------- Kernel 4: y = M @ Q + b_out  (bf16 WMMA, f32 out)
__global__ __launch_bounds__(256) void k_out(const __bf16* __restrict__ qkv,
                                             const __bf16* __restrict__ mpack,
                                             const float* __restrict__ b_out,
                                             float* __restrict__ y) {
  __shared__ __bf16 sB[64 * LDS_PITCH];
  const int nbase = blockIdx.x * 64;
  const int t = threadIdx.x;
  #pragma unroll
  for (int it = 0; it < 32; ++it) {
    int idx = t + it * 256;
    int k = idx >> 6, j = idx & 63;
    sB[j * LDS_PITCH + k] = qkv[k * NPIX + nbase + j];   // q rows are qkv rows 0..127
  }
  __syncthreads();

  const int w = t >> 5, l = t & 31, l16 = l & 15;
  const int nsub = w & 3, grp = w >> 2;
  const int half = (l < 16) ? 0 : 1;

  FragB bfrag[4];
  const __bf16* bp = &sB[(nsub * 16 + l16) * LDS_PITCH + half * 16];
  #pragma unroll
  for (int kc = 0; kc < 4; ++kc) {
    const uint4* p = (const uint4*)(bp + kc * 32);
    bfrag[kc].u[0] = p[0];
    bfrag[kc].u[1] = p[1];
  }
  const int col = nbase + nsub * 16 + l16;

  #pragma unroll
  for (int i = 0; i < 4; ++i) {                 // 8 M-tiles over 2 wave groups
    int mt = i * 2 + grp;
    v8f acc = {};
    #pragma unroll
    for (int kc = 0; kc < 4; ++kc) {
      FragB af;
      const uint4* p = (const uint4*)(mpack + (mt * 4 + kc) * 512 + l * 16);
      af.u[0] = p[0]; af.u[1] = p[1];
      acc = __builtin_amdgcn_wmma_f32_16x16x32_bf16(false, af.v, false, bfrag[kc].v,
                                                    (short)0, acc, false, false);
    }
    int m0 = mt * 16;
    #pragma unroll
    for (int r = 0; r < 8; ++r) {
      int row = m0 + r + half * 8;
      y[row * NPIX + col] = acc[r] + b_out[row];
    }
  }
}

extern "C" void kernel_launch(void* const* d_in, const int* in_sizes, int n_in,
                              void* d_out, int out_size, void* d_ws, size_t ws_size,
                              hipStream_t stream) {
  (void)in_sizes; (void)n_in; (void)out_size; (void)ws_size;
  const float* x     = (const float*)d_in[0];
  const float* w_qkv = (const float*)d_in[1];
  const float* b_qkv = (const float*)d_in[2];
  const float* w_out = (const float*)d_in[3];
  const float* b_out = (const float*)d_in[4];
  float* y = (float*)d_out;

  // workspace map (bytes)
  char* ws = (char*)d_ws;
  __bf16*   qkv   = (__bf16*)ws;                               // 384*147456*2 = 113,246,208
  __bf16*   apack = (__bf16*)(ws + 113246208u);                // 98,304
  __bf16*   mpack = (__bf16*)(ws + 113246208u + 98304u);       // 32,768
  unsigned* kmax  = (unsigned*)(ws + 113246208u + 131072u);    // 512
  float*    ksum  = (float*)(ws + 113246208u + 131584u);       // 512
  float*    ctx   = (float*)(ws + 113246208u + 132096u);       // 16,384

  k_init_pack<<<192, 256, 0, stream>>>(w_qkv, apack, kmax, ksum, ctx);
  k_qkv<<<NPIX / 64, 256, 0, stream>>>(x, apack, b_qkv, qkv, kmax);
  k_ctx<<<288, 128, 0, stream>>>(qkv, kmax, ksum, ctx);
  k_m<<<64, 256, 0, stream>>>(w_out, ctx, ksum, mpack);
  k_out<<<NPIX / 64, 256, 0, stream>>>(qkv, mpack, b_out, y);
}
